// Model_16106127360586
// MI455X (gfx1250) — matrix operand order repaired and checked by reading, not compiled, not run on hardware
//
#include <hip/hip_runtime.h>

// ---------------------------------------------------------------------------
// CDNA5 (gfx1250) hetero-SAGE + link-prediction model.
// - bf16 activations/weights, v_wmma_f32_16x16x32_bf16 for all dense GEMMs
// - CSR-based gather-mean aggregation (u32 atomics only for CSR build)
// - A-fragment reuse across N-tiles; B fragments prefetched per tile
// - accumulate flag templated away (no per-element branches in epilogue)
// ---------------------------------------------------------------------------

typedef __bf16 bf16;
typedef __attribute__((ext_vector_type(16))) __bf16 v16bf;
typedef __attribute__((ext_vector_type(8)))  __bf16 v8bf;
typedef __attribute__((ext_vector_type(8)))  float  v8f;

#define HC       128
#define N_DRUG   20000
#define N_DIS    10000
#define N_ATTR   2000
#define T_ATTR   8
#define L_LAYERS 2
#define E_MT     500000
#define E_HA     100000
#define E_LBL    200000
#define SLOPE    0.01f

__device__ __forceinline__ float leaky_f(float v) { return v >= 0.0f ? v : SLOPE * v; }

// ---- WMMA fragment loaders (layouts per CDNA5 ISA 7.12.2) -------

// A tile 16x32 bf16 from a row pointer.
// lane 0-15 : M=lane,    K = [kbase+0..7]  and [kbase+16..23]
// lane 16-31: M=lane-16, K = [kbase+8..15] and [kbase+24..31]
__device__ __forceinline__ v16bf load_a_row(const bf16* __restrict__ row, int lane, int kbase) {
  int koff = (lane >> 4) << 3;  // 0 or 8
  v8bf lo = *(const v8bf*)(row + kbase + koff);
  v8bf hi = *(const v8bf*)(row + kbase + 16 + koff);
  v16bf r;
#pragma unroll
  for (int i = 0; i < 8; ++i) { r[i] = lo[i]; r[8 + i] = hi[i]; }
  return r;
}

// B tile 32x16 bf16 from row-major W[K][N], leading dim ldb.
// lane l holds column n = n0 + l%16, K = [kbase+koff+0..7] and [kbase+16+koff+0..7], koff = 8*(l/16).
__device__ __forceinline__ v16bf load_b_frag(const bf16* __restrict__ w, int ldb,
                                             int lane, int kbase, int n0) {
  int n = n0 + (lane & 15);
  int koff = (lane >> 4) << 3;
  v16bf r;
#pragma unroll
  for (int i = 0; i < 8; ++i) {
    r[i] = w[(kbase + koff + i) * ldb + n];
    r[8 + i] = w[(kbase + 16 + koff + i) * ldb + n];
  }
  return r;
}

// ---- elementwise / setup kernels -------------------------------------------

__global__ void cvt_f32_bf16_kernel(const float* __restrict__ s, bf16* __restrict__ d, int n) {
  int i = blockIdx.x * blockDim.x + threadIdx.x;
  if (i < n) d[i] = (bf16)s[i];
}

// x_dis = disease_x @ dis_lin_w + dis_lin_b + disease_emb   (no activation)
__global__ void init_dis_kernel(const float* __restrict__ dx, const float* __restrict__ w,
                                const float* __restrict__ b, const float* __restrict__ emb,
                                bf16* __restrict__ xdis_bf) {
  int idx = blockIdx.x * blockDim.x + threadIdx.x;
  if (idx >= N_DIS * HC) return;
  int i = idx >> 7, c = idx & 127;
  float s = b[c] + emb[idx];
#pragma unroll
  for (int k = 0; k < 10; ++k) s += dx[i * 10 + k] * w[k * HC + c];
  xdis_bf[idx] = (bf16)s;
}

__global__ void leaky_bf_kernel(const float* __restrict__ s, bf16* __restrict__ d, int n) {
  int i = blockIdx.x * blockDim.x + threadIdx.x;
  if (i < n) d[i] = (bf16)leaky_f(s[i]);
}

// ---- CSR build (once per call; edge lists are constant) --------------------

__global__ void count_kernel(const int* __restrict__ dst, int* __restrict__ cnt, int E) {
  int e = blockIdx.x * blockDim.x + threadIdx.x;
  if (e < E) atomicAdd(cnt + dst[e], 1);
}

// exclusive scan of cnt[0..n) -> rowptr[0..n], single workgroup
__global__ void scan_kernel(const int* __restrict__ cnt, int* __restrict__ rowptr, int n) {
  __shared__ int partial[256];
  int tid = threadIdx.x;
  int chunk = (n + 255) / 256;
  int beg = tid * chunk;
  int end = beg + chunk; if (end > n) end = n;
  int s = 0;
  for (int i = beg; i < end; ++i) s += cnt[i];
  partial[tid] = s;
  __syncthreads();
  if (tid == 0) {
    int run = 0;
    for (int i = 0; i < 256; ++i) { int v = partial[i]; partial[i] = run; run += v; }
    rowptr[n] = run;
  }
  __syncthreads();
  int run = partial[tid];
  for (int i = beg; i < end; ++i) { rowptr[i] = run; run += cnt[i]; }
}

// bin source node ids by destination: ebuf[cursor[dst[e]]++] = src[e]
__global__ void fill_csr_kernel(const int* __restrict__ src, const int* __restrict__ dst,
                                int* __restrict__ cursor, int* __restrict__ ebuf, int E) {
  int e = blockIdx.x * blockDim.x + threadIdx.x;
  if (e >= E) return;
  int pos = atomicAdd(cursor + dst[e], 1);
  ebuf[pos] = src[e];
}

// ---- gather-mean aggregation (no f32 atomics) ------------------------------
// 16 threads per node, 8 features each; mean of neighbor rows -> bf16
__global__ void gather_mean_kernel(const bf16* __restrict__ feat, const int* __restrict__ ebuf,
                                   const int* __restrict__ rowptr, bf16* __restrict__ mean_bf,
                                   int n_nodes) {
  int t = blockIdx.x * blockDim.x + threadIdx.x;
  int node = t >> 4;
  if (node >= n_nodes) return;
  int c = (t & 15) << 3;
  int beg = rowptr[node], end = rowptr[node + 1];
  float acc[8] = {0.f, 0.f, 0.f, 0.f, 0.f, 0.f, 0.f, 0.f};
  for (int i = beg; i < end; ++i) {
    v8bf v = *(const v8bf*)(feat + ebuf[i] * HC + c);
#pragma unroll
    for (int j = 0; j < 8; ++j) acc[j] += (float)v[j];
  }
  float inv = 1.0f / fmaxf((float)(end - beg), 1.0f);
  bf16* orow = mean_bf + node * HC + c;
#pragma unroll
  for (int j = 0; j < 8; ++j) orow[j] = (bf16)(acc[j] * inv);
}

// ---- WMMA GEMMs ------------------------------------------------------------

// out[m,:] (+)= meanA[m,:] @ wl + xdst[m,:] @ wr + bl    (f32 out)
// one wave per 16-row block; A fragments loaded once, loop over 8 N-tiles,
// B fragments for the whole tile prefetched before the WMMA chain.
template <int ACC>
__global__ void __launch_bounds__(32) sage_gemm_kernel(
    const bf16* __restrict__ meanA, const bf16* __restrict__ xdst,
    const bf16* __restrict__ wl, const bf16* __restrict__ wr,
    const float* __restrict__ bl, float* __restrict__ out) {
  int lane = threadIdx.x;
  int m0 = blockIdx.x << 4;
  const bf16* arow1 = meanA + (m0 + (lane & 15)) * HC;
  const bf16* arow2 = xdst + (m0 + (lane & 15)) * HC;
  v16bf a1[4], a2[4];
#pragma unroll
  for (int ks = 0; ks < 4; ++ks) {
    a1[ks] = load_a_row(arow1, lane, ks << 5);
    a2[ks] = load_a_row(arow2, lane, ks << 5);
  }
  for (int tn = 0; tn < 8; ++tn) {
    int n0 = tn << 4;
    v16bf b1[4], b2[4];
#pragma unroll
    for (int ks = 0; ks < 4; ++ks) {
      b1[ks] = load_b_frag(wl, HC, lane, ks << 5, n0);
      b2[ks] = load_b_frag(wr, HC, lane, ks << 5, n0);
    }
    float bias = bl[n0 + (lane & 15)];
    float* op = out + (m0 + ((lane >> 4) << 3)) * HC + n0 + (lane & 15);
    float prev[8];
    if (ACC) {
#pragma unroll
      for (int i = 0; i < 8; ++i) prev[i] = op[i * HC];
    }
    v8f acc = {0.f, 0.f, 0.f, 0.f, 0.f, 0.f, 0.f, 0.f};
#pragma unroll
    for (int ks = 0; ks < 4; ++ks) {
      acc = __builtin_amdgcn_wmma_f32_16x16x32_bf16(false, a1[ks], false, b1[ks], (short)0, acc, false, false);
      acc = __builtin_amdgcn_wmma_f32_16x16x32_bf16(false, a2[ks], false, b2[ks], (short)0, acc, false, false);
    }
#pragma unroll
    for (int i = 0; i < 8; ++i) {
      float v = acc[i] + bias;
      if (ACC) v += prev[i];
      op[i * HC] = v;
    }
  }
}

// classifier stage 1: h0 = leaky( concat(x_drug[src], x_dis[dst]) @ cw0 + cb0 )
// gather fused into A-fragment loads (done once), loop over 4 N-tiles with
// per-tile B prefetch.
__global__ void __launch_bounds__(32) cls_gemm_kernel(
    const bf16* __restrict__ xdrug, const bf16* __restrict__ xdis,
    const int* __restrict__ esrc, const int* __restrict__ edst,
    const bf16* __restrict__ w0, const float* __restrict__ b0,
    float* __restrict__ h0) {
  int lane = threadIdx.x;
  int m0 = blockIdx.x << 4;
  int r = m0 + (lane & 15);
  const bf16* rowd = xdrug + esrc[r] * HC;
  const bf16* rowi = xdis + edst[r] * HC;
  v16bf a[8];
#pragma unroll
  for (int ks = 0; ks < 8; ++ks)
    a[ks] = load_a_row((ks < 4) ? rowd : rowi, lane, (ks & 3) << 5);
  for (int tn = 0; tn < 4; ++tn) {
    int n0 = tn << 4;
    v16bf b[8];
#pragma unroll
    for (int ks = 0; ks < 8; ++ks) b[ks] = load_b_frag(w0, 64, lane, ks << 5, n0);
    v8f acc = {0.f, 0.f, 0.f, 0.f, 0.f, 0.f, 0.f, 0.f};
#pragma unroll
    for (int ks = 0; ks < 8; ++ks)
      acc = __builtin_amdgcn_wmma_f32_16x16x32_bf16(false, a[ks], false, b[ks], (short)0, acc, false, false);
    float bias = b0[n0 + (lane & 15)];
    float* op = h0 + (m0 + ((lane >> 4) << 3)) * 64 + n0 + (lane & 15);
#pragma unroll
    for (int i = 0; i < 8; ++i) op[i * 64] = leaky_f(acc[i] + bias);
  }
}

// tail MLP: 64 -> 32 -> 16 -> 8 -> 1 (tiny; one thread per edge)
__global__ void mlp_tail_kernel(const float* __restrict__ h0,
                                const float* __restrict__ cw1, const float* __restrict__ cb1,
                                const float* __restrict__ cw2, const float* __restrict__ cb2,
                                const float* __restrict__ cw3, const float* __restrict__ cb3,
                                const float* __restrict__ cw4, const float* __restrict__ cb4,
                                float* __restrict__ out, int n) {
  int e = blockIdx.x * blockDim.x + threadIdx.x;
  if (e >= n) return;
  const float* h = h0 + e * 64;
  float h1[32];
  for (int j = 0; j < 32; ++j) {
    float s = cb1[j];
    for (int i = 0; i < 64; ++i) s += h[i] * cw1[i * 32 + j];
    h1[j] = leaky_f(s);
  }
  float h2[16];
  for (int j = 0; j < 16; ++j) {
    float s = cb2[j];
    for (int i = 0; i < 32; ++i) s += h1[i] * cw2[i * 16 + j];
    h2[j] = leaky_f(s);
  }
  float h3[8];
  for (int j = 0; j < 8; ++j) {
    float s = cb3[j];
    for (int i = 0; i < 16; ++i) s += h2[i] * cw3[i * 8 + j];
    h3[j] = leaky_f(s);
  }
  float pv = cb4[0];
  for (int i = 0; i < 8; ++i) pv += h3[i] * cw4[i];
  out[e] = pv;
}

// ---------------------------------------------------------------------------

extern "C" void kernel_launch(void* const* d_in, const int* in_sizes, int n_in,
                              void* d_out, int out_size, void* d_ws, size_t ws_size,
                              hipStream_t stream) {
  (void)in_sizes; (void)n_in; (void)out_size;

  const float* disease_x   = (const float*)d_in[0];
  const float* drug_emb    = (const float*)d_in[1];
  const float* disease_emb = (const float*)d_in[2];
  const float* attr_emb    = (const float*)d_in[3];
  const float* dis_lin_w   = (const float*)d_in[4];
  const float* dis_lin_b   = (const float*)d_in[5];
  const float* mt_wl  = (const float*)d_in[6];
  const float* mt_bl  = (const float*)d_in[7];
  const float* mt_wr  = (const float*)d_in[8];
  const float* rmt_wl = (const float*)d_in[9];
  const float* rmt_bl = (const float*)d_in[10];
  const float* rmt_wr = (const float*)d_in[11];
  const float* ha_wl  = (const float*)d_in[12];
  const float* ha_bl  = (const float*)d_in[13];
  const float* ha_wr  = (const float*)d_in[14];
  const float* rha_wl = (const float*)d_in[15];
  const float* rha_bl = (const float*)d_in[16];
  const float* rha_wr = (const float*)d_in[17];
  const float* cw0 = (const float*)d_in[18];
  const float* cb0 = (const float*)d_in[19];
  const float* cw1 = (const float*)d_in[20];
  const float* cb1 = (const float*)d_in[21];
  const float* cw2 = (const float*)d_in[22];
  const float* cb2 = (const float*)d_in[23];
  const float* cw3 = (const float*)d_in[24];
  const float* cb3 = (const float*)d_in[25];
  const float* cw4 = (const float*)d_in[26];
  const float* cb4 = (const float*)d_in[27];
  const int* mt_src  = (const int*)d_in[28];
  const int* mt_dst  = (const int*)d_in[29];
  const int* ha_src  = (const int*)d_in[30];
  const int* ha_dst  = (const int*)d_in[31];
  const int* ell_src = (const int*)d_in[32];
  const int* ell_dst = (const int*)d_in[33];
  float* out = (float*)d_out;

  // ---- workspace carve (256B aligned chunks) ----
  char* p = (char*)d_ws;
  auto carve = [&](size_t bytes) -> void* {
    void* r = (void*)p;
    p += (bytes + 255) & ~(size_t)255;
    return r;
  };
  bf16* x_drug_bf    = (bf16*)carve((size_t)N_DRUG * HC * 2);
  bf16* x_dis_bf     = (bf16*)carve((size_t)N_DIS * HC * 2);
  bf16* x_attr_bf    = (bf16*)carve((size_t)T_ATTR * N_ATTR * HC * 2);
  bf16* mean_dis_bf  = (bf16*)carve((size_t)N_DIS * HC * 2);
  bf16* mean_drug_bf = (bf16*)carve((size_t)N_DRUG * HC * 2);
  bf16* mean_attr_bf = (bf16*)carve((size_t)T_ATTR * N_ATTR * HC * 2);
  bf16* mt_wl_bf  = (bf16*)carve((size_t)L_LAYERS * HC * HC * 2);
  bf16* mt_wr_bf  = (bf16*)carve((size_t)L_LAYERS * HC * HC * 2);
  bf16* rmt_wl_bf = (bf16*)carve((size_t)L_LAYERS * HC * HC * 2);
  bf16* rmt_wr_bf = (bf16*)carve((size_t)L_LAYERS * HC * HC * 2);
  bf16* ha_wl_bf  = (bf16*)carve((size_t)L_LAYERS * T_ATTR * HC * HC * 2);
  bf16* ha_wr_bf  = (bf16*)carve((size_t)L_LAYERS * T_ATTR * HC * HC * 2);
  bf16* rha_wl_bf = (bf16*)carve((size_t)L_LAYERS * T_ATTR * HC * HC * 2);
  bf16* rha_wr_bf = (bf16*)carve((size_t)L_LAYERS * T_ATTR * HC * HC * 2);
  bf16* cw0_bf    = (bf16*)carve((size_t)2 * HC * 64 * 2);
  float* dis_next  = (float*)carve((size_t)N_DIS * HC * 4);
  float* attr_next = (float*)carve((size_t)T_ATTR * N_ATTR * HC * 4);
  float* drug_acc  = (float*)carve((size_t)N_DRUG * HC * 4);
  // CSR state (built once per call)
  int* cnt_dis     = (int*)carve((size_t)N_DIS * 4);               // also reused as cursor
  int* cnt_drug_mt = (int*)carve((size_t)N_DRUG * 4);
  int* cnt_attr    = (int*)carve((size_t)T_ATTR * N_ATTR * 4);
  int* cnt_drug_ha = (int*)carve((size_t)T_ATTR * N_DRUG * 4);
  int* rp_dis      = (int*)carve((size_t)(N_DIS + 1) * 4);
  int* rp_drug_mt  = (int*)carve((size_t)(N_DRUG + 1) * 4);
  int* rp_attr     = (int*)carve((size_t)T_ATTR * (N_ATTR + 1) * 4);
  int* rp_drug_ha  = (int*)carve((size_t)T_ATTR * (N_DRUG + 1) * 4);
  int* eb_dis      = (int*)carve((size_t)E_MT * 4);
  int* eb_drug_mt  = (int*)carve((size_t)E_MT * 4);
  int* eb_attr     = (int*)carve((size_t)T_ATTR * E_HA * 4);
  int* eb_drug_ha  = (int*)carve((size_t)T_ATTR * E_HA * 4);
  float* h0 = (float*)carve((size_t)E_LBL * 64 * 4);
  if ((size_t)(p - (char*)d_ws) > ws_size) return;  // insufficient scratch

  auto cvt = [&](const float* s, bf16* d, int n) {
    cvt_f32_bf16_kernel<<<(n + 255) / 256, 256, 0, stream>>>(s, d, n);
  };

  // ---- activations / weights to bf16 ----
  cvt(drug_emb, x_drug_bf, N_DRUG * HC);
  cvt(attr_emb, x_attr_bf, T_ATTR * N_ATTR * HC);
  init_dis_kernel<<<(N_DIS * HC + 255) / 256, 256, 0, stream>>>(
      disease_x, dis_lin_w, dis_lin_b, disease_emb, x_dis_bf);
  cvt(mt_wl,  mt_wl_bf,  L_LAYERS * HC * HC);
  cvt(mt_wr,  mt_wr_bf,  L_LAYERS * HC * HC);
  cvt(rmt_wl, rmt_wl_bf, L_LAYERS * HC * HC);
  cvt(rmt_wr, rmt_wr_bf, L_LAYERS * HC * HC);
  cvt(ha_wl,  ha_wl_bf,  L_LAYERS * T_ATTR * HC * HC);
  cvt(ha_wr,  ha_wr_bf,  L_LAYERS * T_ATTR * HC * HC);
  cvt(rha_wl, rha_wl_bf, L_LAYERS * T_ATTR * HC * HC);
  cvt(rha_wr, rha_wr_bf, L_LAYERS * T_ATTR * HC * HC);
  cvt(cw0,    cw0_bf,    2 * HC * 64);

  // ---- CSR build (counts -> scan -> bin), once per call ----
  auto build_csr = [&](const int* src, const int* dst, int E, int n_nodes,
                       int* cnt, int* rp, int* eb) {
    hipMemsetAsync(cnt, 0, (size_t)n_nodes * 4, stream);
    count_kernel<<<(E + 255) / 256, 256, 0, stream>>>(dst, cnt, E);
    scan_kernel<<<1, 256, 0, stream>>>(cnt, rp, n_nodes);
    hipMemcpyAsync(cnt, rp, (size_t)n_nodes * 4, hipMemcpyDeviceToDevice, stream);  // cursor
    fill_csr_kernel<<<(E + 255) / 256, 256, 0, stream>>>(src, dst, cnt, eb, E);
  };
  build_csr(mt_src, mt_dst, E_MT, N_DIS, cnt_dis, rp_dis, eb_dis);
  build_csr(mt_dst, mt_src, E_MT, N_DRUG, cnt_drug_mt, rp_drug_mt, eb_drug_mt);
  for (int t = 0; t < T_ATTR; ++t) {
    build_csr(ha_src + t * E_HA, ha_dst + t * E_HA, E_HA, N_ATTR,
              cnt_attr + t * N_ATTR, rp_attr + t * (N_ATTR + 1), eb_attr + t * E_HA);
    build_csr(ha_dst + t * E_HA, ha_src + t * E_HA, E_HA, N_DRUG,
              cnt_drug_ha + t * N_DRUG, rp_drug_ha + t * (N_DRUG + 1), eb_drug_ha + t * E_HA);
  }

  auto gather_mean = [&](const bf16* feat, const int* eb, const int* rp, bf16* mean, int n_nodes) {
    gather_mean_kernel<<<(n_nodes * 16 + 255) / 256, 256, 0, stream>>>(feat, eb, rp, mean, n_nodes);
  };

  // ---- GNN layers ----
  for (int l = 0; l < L_LAYERS; ++l) {
    const bf16* Wmtl  = mt_wl_bf  + l * HC * HC;
    const bf16* Wmtr  = mt_wr_bf  + l * HC * HC;
    const bf16* Wrmtl = rmt_wl_bf + l * HC * HC;
    const bf16* Wrmtr = rmt_wr_bf + l * HC * HC;

    // drug -> disease (may_treat)
    gather_mean(x_drug_bf, eb_dis, rp_dis, mean_dis_bf, N_DIS);
    sage_gemm_kernel<0><<<N_DIS / 16, 32, 0, stream>>>(
        mean_dis_bf, x_dis_bf, Wmtl, Wmtr, mt_bl + l * HC, dis_next);

    // disease -> drug (rev_may_treat): first drug contribution, overwrite
    gather_mean(x_dis_bf, eb_drug_mt, rp_drug_mt, mean_drug_bf, N_DRUG);
    sage_gemm_kernel<0><<<N_DRUG / 16, 32, 0, stream>>>(
        mean_drug_bf, x_drug_bf, Wrmtl, Wrmtr, rmt_bl + l * HC, drug_acc);

    // attribute relations
    for (int t = 0; t < T_ATTR; ++t) {
      const bf16* xa = x_attr_bf + (size_t)t * N_ATTR * HC;
      bf16* mean_a = mean_attr_bf + (size_t)t * N_ATTR * HC;
      int wi = (l * T_ATTR + t) * HC * HC;
      int bi = (l * T_ATTR + t) * HC;

      // drug -> attr_t (has)
      gather_mean(x_drug_bf, eb_attr + t * E_HA, rp_attr + t * (N_ATTR + 1), mean_a, N_ATTR);
      sage_gemm_kernel<0><<<N_ATTR / 16, 32, 0, stream>>>(
          mean_a, xa, ha_wl_bf + wi, ha_wr_bf + wi, ha_bl + bi,
          attr_next + (size_t)t * N_ATTR * HC);

      // attr_t -> drug (rev_has), accumulate into drug_acc
      gather_mean(xa, eb_drug_ha + t * E_HA, rp_drug_ha + t * (N_DRUG + 1), mean_drug_bf, N_DRUG);
      sage_gemm_kernel<1><<<N_DRUG / 16, 32, 0, stream>>>(
          mean_drug_bf, x_drug_bf, rha_wl_bf + wi, rha_wr_bf + wi, rha_bl + bi, drug_acc);
    }

    // epilogue: leaky ReLU -> next-layer bf16 activations
    leaky_bf_kernel<<<(N_DIS * HC + 255) / 256, 256, 0, stream>>>(dis_next, x_dis_bf, N_DIS * HC);
    leaky_bf_kernel<<<(T_ATTR * N_ATTR * HC + 255) / 256, 256, 0, stream>>>(attr_next, x_attr_bf, T_ATTR * N_ATTR * HC);
    leaky_bf_kernel<<<(N_DRUG * HC + 255) / 256, 256, 0, stream>>>(drug_acc, x_drug_bf, N_DRUG * HC);
  }

  // ---- classifier ----
  cls_gemm_kernel<<<E_LBL / 16, 32, 0, stream>>>(
      x_drug_bf, x_dis_bf, ell_src, ell_dst, cw0_bf, cb0, h0);
  mlp_tail_kernel<<<(E_LBL + 127) / 128, 128, 0, stream>>>(
      h0, cw1, cb1, cw2, cb2, cw3, cb3, cw4, cb4, out, E_LBL);
}
